// SortedSetTransformerAggregation_16758962389562
// MI455X (gfx1250) — matrix operand order repaired and checked by reading, not compiled, not run on hardware
//
#include <hip/hip_runtime.h>
#include <stdint.h>

#define C_    128
#define CO_   256
#define B_    2048
#define N_    131072
#define LMAX_ 128
#define M_    32

typedef __attribute__((ext_vector_type(16))) __bf16       v16bf;
typedef __attribute__((ext_vector_type(8)))  float        v8f;
typedef __attribute__((ext_vector_type(4)))  unsigned int u32x4;
typedef __attribute__((ext_vector_type(8)))  int          i32x8;
typedef __attribute__((ext_vector_type(4)))  int          i32x4;

#if defined(__HIP_DEVICE_COMPILE__)
#if __has_builtin(__builtin_amdgcn_tensor_load_to_lds) && __has_builtin(__builtin_amdgcn_s_wait_tensorcnt)
#define USE_TDM 1
#endif
#endif

// ---------------- WMMA operand helpers (layouts per CDNA5 ISA 7.12.2) ----------------

__device__ __forceinline__ int imin(int a, int b) { return a < b ? a : b; }

// A operand, bf16 16x32 (MxK): lane<16 holds row=lane, K = k0+{0..7,16..23};
// lane>=16 holds row=lane-16, K = k0+{8..15,24..31}.
__device__ __forceinline__ v16bf load_a_bf(const __bf16* base, int stride, int r0, int k0, int lane) {
  int m = lane & 15;
  const __bf16* p = base + (r0 + m) * stride + k0 + ((lane & 16) ? 8 : 0);
  v16bf a;
#pragma unroll
  for (int i = 0; i < 8; ++i) { a[i] = p[i]; a[i + 8] = p[i + 16]; }
  return a;
}

// A operand sourced from f32 (softmax probs), converted to bf16 in-register.
__device__ __forceinline__ v16bf load_a_f32(const float* base, int stride, int r0, int k0, int lane) {
  int m = lane & 15;
  const float* p = base + (r0 + m) * stride + k0 + ((lane & 16) ? 8 : 0);
  v16bf a;
#pragma unroll
  for (int i = 0; i < 8; ++i) { a[i] = (__bf16)p[i]; a[i + 8] = (__bf16)p[i + 16]; }
  return a;
}

// B operand, bf16 32x16 (KxN), sourced from a row-major matrix Mrow[n][k]
// (i.e. B = Mrow^T): lane<16 holds col n=lane, K=k0+{0..7,16..23}; lane>=16: K=k0+{8..15,24..31}.
__device__ __forceinline__ v16bf load_b_bf(const __bf16* base, int stride, int n0, int k0, int lane) {
  int n = lane & 15;
  const __bf16* p = base + (n0 + n) * stride + k0 + ((lane & 16) ? 8 : 0);
  v16bf b;
#pragma unroll
  for (int i = 0; i < 8; ++i) { b[i] = p[i]; b[i + 8] = p[i + 16]; }
  return b;
}

// C/D layout: lane<16: col=lane, rows r0+0..7 ; lane>=16: col=lane-16, rows r0+8..15.
__device__ __forceinline__ void store_d_bf(__bf16* base, int stride, int r0, int c0, int lane, v8f d, float bias) {
  int n = lane & 15, mb = (lane & 16) ? 8 : 0;
#pragma unroll
  for (int r = 0; r < 8; ++r) base[(r0 + mb + r) * stride + c0 + n] = (__bf16)(d[r] + bias);
}
__device__ __forceinline__ void store_dT_bf(__bf16* base, int stride, int r0, int c0, int lane, v8f d, float bias) {
  int n = lane & 15, mb = (lane & 16) ? 8 : 0;
#pragma unroll
  for (int r = 0; r < 8; ++r) base[(c0 + n) * stride + r0 + mb + r] = (__bf16)(d[r] + bias);
}
__device__ __forceinline__ void store_d_f32(float* base, int stride, int r0, int c0, int lane, v8f d, float bias) {
  int n = lane & 15, mb = (lane & 16) ? 8 : 0;
#pragma unroll
  for (int r = 0; r < 8; ++r) base[(r0 + mb + r) * stride + c0 + n] = d[r] + bias;
}

#define WMMA_BF16(a, b, c) \
  __builtin_amdgcn_wmma_f32_16x16x32_bf16(false, (a), false, (b), (short)0, (c), false, false)

__device__ __forceinline__ float wave_sum(float v) {
#pragma unroll
  for (int o = 16; o; o >>= 1) v += __shfl_xor(v, o, 32);
  return v;
}
__device__ __forceinline__ float wave_max(float v) {
#pragma unroll
  for (int o = 16; o; o >>= 1) v = fmaxf(v, __shfl_xor(v, o, 32));
  return v;
}

// ---------------- TDM: 2D tile (rows x 128 bf16) global -> LDS, pad 8 bf16/row ----------------
#ifdef USE_TDM
__device__ __forceinline__ void tdm_load_tile(unsigned lds_off, const void* gptr, int rows) {
  unsigned long long ga = (unsigned long long)(uintptr_t)gptr;
  u32x4 g0;
  g0[0] = 1u;                                     // count=1, user descriptor, no gather
  g0[1] = lds_off;                                // lds_addr (bytes)
  g0[2] = (unsigned)(ga & 0xffffffffu);           // global_addr[31:0]
  g0[3] = (unsigned)((ga >> 32) & 0x1ffffffu) | (2u << 30);  // global_addr[56:32] | type=2
  i32x8 g1;
  // data_size=1 (2B) | pad_enable | pad_interval=5 (64 DW = one 128-elem row) | pad_amount=3 (4 DW = 8 bf16)
  g1[0] = (1 << 16) | (1 << 20) | (5 << 22) | (3 << 25);
  g1[1] = (128 << 16);                            // tensor_dim0 = 128 (low 16 at [31:16])
  g1[2] = ((rows & 0xffff) << 16);                // tensor_dim0 hi = 0 ; tensor_dim1 low16
  g1[3] = (128 << 16);                            // tensor_dim1 hi = 0 ; tile_dim0 = 128
  g1[4] = (rows & 0xffff);                        // tile_dim1 = rows ; tile_dim2 = 0
  g1[5] = 128;                                    // tensor_dim0_stride (low 32)
  g1[6] = 0;
  g1[7] = 0;
  i32x4 z4 = {0, 0, 0, 0};
  i32x8 z8 = {0, 0, 0, 0, 0, 0, 0, 0};
  __builtin_amdgcn_tensor_load_to_lds(g0, g1, z4, z4, z8, 0);
}
#endif

// ---------------- small kernels ----------------

__global__ __launch_bounds__(256) void k_starts(const int* __restrict__ index,
                                                int* __restrict__ starts, int* __restrict__ counts) {
  int b = blockIdx.x * blockDim.x + threadIdx.x;
  if (b >= B_) return;
  int lo = 0, hi = N_;
  while (lo < hi) { int m = (lo + hi) >> 1; if (index[m] < b) lo = m + 1; else hi = m; }
  int s = lo;
  lo = 0; hi = N_;
  while (lo < hi) { int m = (lo + hi) >> 1; if (index[m] < b + 1) lo = m + 1; else hi = m; }
  starts[b] = s;
  int c = lo - s;
  counts[b] = c > LMAX_ ? LMAX_ : c;
}

__global__ __launch_bounds__(256) void k_gather(const float* __restrict__ x,
                                                const int* __restrict__ starts,
                                                const int* __restrict__ counts,
                                                __bf16* __restrict__ dense) {
  int row = blockIdx.x * 8 + (threadIdx.x >> 5);
  int lane = threadIdx.x & 31;
  int b = row >> 7, l = row & 127;
  int cnt = counts[b];
  long long dst = (long long)row * C_;
  if (l < cnt) {
    long long src = ((long long)starts[b] + l) * C_;
#pragma unroll
    for (int j = 0; j < 4; ++j) dense[dst + lane + 32 * j] = (__bf16)x[src + lane + 32 * j];
  } else {
#pragma unroll
    for (int j = 0; j < 4; ++j) dense[dst + lane + 32 * j] = (__bf16)0.0f;
  }
}

__global__ __launch_bounds__(256) void k_cvt(const float* __restrict__ src,
                                             __bf16* __restrict__ dst, int n) {
  int i = blockIdx.x * 256 + threadIdx.x;
  if (i < n) dst[i] = (__bf16)src[i];
}

// ---------------- generic WMMA linear: out[M,N] = act(A[M,128] @ W[N,128]^T + bias) ----------------

struct LinP {
  const __bf16* A;
  const __bf16* W;
  const float*  bias;
  int N, relu, outf32;
  void* out;
};

__global__ __launch_bounds__(256) void k_linear(LinP P) {
  __shared__ __bf16 As[128 * 136];
  const int tid = threadIdx.x, lane = tid & 31, wv = tid >> 5;
  const long long r0 = (long long)blockIdx.x * 128;

  for (int r = wv; r < 128; r += 8) {
    const __bf16* src = P.A + (r0 + r) * C_;
#pragma unroll
    for (int j = 0; j < 4; ++j) As[r * 136 + lane + 32 * j] = src[lane + 32 * j];
  }
  __syncthreads();

  const int ctiles = P.N >> 4;
  for (int t = wv; t < 8 * ctiles; t += 8) {
    int tr = t / ctiles, tc = t % ctiles;
    v8f c = {};
#pragma unroll
    for (int k0 = 0; k0 < 128; k0 += 32) {
      v16bf a = load_a_bf(As, 136, tr * 16, k0, lane);
      v16bf b = load_b_bf(P.W, 128, tc * 16, k0, lane);
      c = WMMA_BF16(a, b, c);
    }
    int n = lane & 15, mb = (lane & 16) ? 8 : 0;
    float bias = P.bias[tc * 16 + n];
#pragma unroll
    for (int r = 0; r < 8; ++r) {
      float v = c[r] + bias;
      if (P.relu) v = fmaxf(v, 0.0f);
      long long row = r0 + tr * 16 + mb + r;
      long long col = tc * 16 + n;
      if (P.outf32) ((float*)P.out)[row * P.N + col] = v;
      else          ((__bf16*)P.out)[row * P.N + col] = (__bf16)v;
    }
  }
}

// ---------------- fused MultiheadAttentionBlock (one block per batch element) ----------------

struct MabP {
  const __bf16* X; long long xBS; int Lq, xRows, LqStore;
  const __bf16* Y; long long yBS; int Lk, yRows;
  int use_counts, kfix, xmask;
  const __bf16 *Wq, *Wk, *Wv, *Wo, *Wl;
  const float *bq, *bk, *bv, *bo, *bl, *g1, *b1, *g2, *b2;
  const int* counts;
  __bf16* out; long long oBS;
};

// LDS: Xs[128][136] Ys[128][136] Os[128][136] bf16; Qh[128][40] Kh[128][40] bf16;
//      Vt[32][136] bf16 (V transposed); Sf[128][132] f32 (scores/probs, then row buffer)
#define MAB_LDS_BYTES (((3 * 128 * 136) + (2 * 128 * 40) + (32 * 136)) * 2 + 128 * 132 * 4)

__global__ __launch_bounds__(256) void k_mab(MabP P) {
  extern __shared__ char smem_raw[];
  __bf16* Xs = (__bf16*)smem_raw;
  __bf16* Ys = Xs + 128 * 136;
  __bf16* Os = Ys + 128 * 136;
  __bf16* Qh = Os + 128 * 136;
  __bf16* Kh = Qh + 128 * 40;
  __bf16* Vt = Kh + 128 * 40;
  float*  Sf = (float*)(Vt + 32 * 136);

  const int tid = threadIdx.x, lane = tid & 31, wv = tid >> 5;
  const int b = blockIdx.x;
  const int Lq = P.Lq, Lk = P.Lk;
  const int LqT = Lq >> 4, LkT = Lk >> 4;

  const int cnt    = P.counts[b];
  const int kcount = P.use_counts ? imin(cnt, Lk) : P.kfix;
  const int qcnt   = imin(cnt, Lq);

  if (tid == 0) {
    __builtin_prefetch(P.Wq, 0, 1); __builtin_prefetch(P.Wk, 0, 1);
    __builtin_prefetch(P.Wv, 0, 1); __builtin_prefetch(P.Wo, 0, 1);
    __builtin_prefetch(P.Wl, 0, 1);
  }

  // zero the padding rows (so padded K/V rows stay finite and A-tiles are clean)
  for (int r = P.xRows + wv; r < Lq; r += 8)
#pragma unroll
    for (int j = 0; j < 4; ++j) Xs[r * 136 + lane + 32 * j] = (__bf16)0.0f;
  for (int r = P.yRows + wv; r < Lk; r += 8)
#pragma unroll
    for (int j = 0; j < 4; ++j) Ys[r * 136 + lane + 32 * j] = (__bf16)0.0f;

  const __bf16* Xg = P.X + (long long)b * P.xBS;
  const __bf16* Yg = P.Y + (long long)b * P.yBS;

#ifdef USE_TDM
  if (wv == 0) {
    tdm_load_tile((unsigned)(uintptr_t)Xs, Xg, P.xRows);
    tdm_load_tile((unsigned)(uintptr_t)Ys, Yg, P.yRows);
    __builtin_amdgcn_s_wait_tensorcnt(0);
  }
#else
  for (int r = wv; r < P.xRows; r += 8)
#pragma unroll
    for (int j = 0; j < 4; ++j) Xs[r * 136 + lane + 32 * j] = Xg[(long long)r * C_ + lane + 32 * j];
  for (int r = wv; r < P.yRows; r += 8)
#pragma unroll
    for (int j = 0; j < 4; ++j) Ys[r * 136 + lane + 32 * j] = Yg[(long long)r * C_ + lane + 32 * j];
#endif
  __syncthreads();

  const float scale = 0.17677669529663688f;  // 1/sqrt(32)

  // ---------------- attention: per-head ----------------
  for (int h = 0; h < 4; ++h) {
    // Q_h = Xs @ Wq_h^T + bq_h  -> Qh [Lq,32]
    for (int t = wv; t < LqT * 2; t += 8) {
      int tr = t >> 1, tc = t & 1;
      v8f c = {};
#pragma unroll
      for (int k0 = 0; k0 < 128; k0 += 32) {
        v16bf a = load_a_bf(Xs, 136, tr * 16, k0, lane);
        v16bf bb = load_b_bf(P.Wq, 128, h * 32 + tc * 16, k0, lane);
        c = WMMA_BF16(a, bb, c);
      }
      store_d_bf(Qh, 40, tr * 16, tc * 16, lane, c, P.bq[h * 32 + tc * 16 + (lane & 15)]);
    }
    // K_h = Ys @ Wk_h^T + bk_h -> Kh [Lk,32]
    for (int t = wv; t < LkT * 2; t += 8) {
      int tr = t >> 1, tc = t & 1;
      v8f c = {};
#pragma unroll
      for (int k0 = 0; k0 < 128; k0 += 32) {
        v16bf a = load_a_bf(Ys, 136, tr * 16, k0, lane);
        v16bf bb = load_b_bf(P.Wk, 128, h * 32 + tc * 16, k0, lane);
        c = WMMA_BF16(a, bb, c);
      }
      store_d_bf(Kh, 40, tr * 16, tc * 16, lane, c, P.bk[h * 32 + tc * 16 + (lane & 15)]);
    }
    // V_h = Ys @ Wv_h^T + bv_h -> stored transposed Vt [32][Lk]
    for (int t = wv; t < LkT * 2; t += 8) {
      int tr = t >> 1, tc = t & 1;
      v8f c = {};
#pragma unroll
      for (int k0 = 0; k0 < 128; k0 += 32) {
        v16bf a = load_a_bf(Ys, 136, tr * 16, k0, lane);
        v16bf bb = load_b_bf(P.Wv, 128, h * 32 + tc * 16, k0, lane);
        c = WMMA_BF16(a, bb, c);
      }
      store_dT_bf(Vt, 136, tr * 16, tc * 16, lane, c, P.bv[h * 32 + tc * 16 + (lane & 15)]);
    }
    __syncthreads();

    // scores S = Qh @ Kh^T * scale -> Sf [Lq][Lk] f32   (single K=32 WMMA per tile)
    for (int t = wv; t < LqT * LkT; t += 8) {
      int tr = t / LkT, tc = t % LkT;
      v8f c = {};
      v16bf a = load_a_bf(Qh, 40, tr * 16, 0, lane);
      v16bf bb = load_b_bf(Kh, 40, tc * 16, 0, lane);
      c = WMMA_BF16(a, bb, c);
      int n = lane & 15, mb = (lane & 16) ? 8 : 0;
#pragma unroll
      for (int r = 0; r < 8; ++r)
        Sf[(tr * 16 + mb + r) * 132 + tc * 16 + n] = c[r] * scale;
    }
    __syncthreads();

    // masked softmax rows (kt >= kcount -> 0, matching the -1e9 mask)
    for (int q = wv; q < Lq; q += 8) {
      float m = -3.0e38f;
      for (int kt = lane; kt < Lk; kt += 32)
        if (kt < kcount) m = fmaxf(m, Sf[q * 132 + kt]);
      m = wave_max(m);
      float s = 0.0f;
      for (int kt = lane; kt < Lk; kt += 32) {
        float p = (kt < kcount) ? __expf(Sf[q * 132 + kt] - m) : 0.0f;
        Sf[q * 132 + kt] = p;
        s += p;
      }
      s = wave_sum(s);
      float inv = 1.0f / s;
      for (int kt = lane; kt < Lk; kt += 32) Sf[q * 132 + kt] *= inv;
    }
    __syncthreads();

    // O[:, h*32:h*32+32] = P @ V_h  (A = probs f32->bf16, B = Vt rows contiguous)
    for (int t = wv; t < LqT * 2; t += 8) {
      int tr = t >> 1, tc = t & 1;
      v8f c = {};
      for (int k0 = 0; k0 < Lk; k0 += 32) {
        v16bf a = load_a_f32(Sf, 132, tr * 16, k0, lane);
        v16bf bb = load_b_bf(Vt, 136, tc * 16, k0, lane);
        c = WMMA_BF16(a, bb, c);
      }
      store_d_bf(Os, 136, tr * 16, h * 32 + tc * 16, lane, c, 0.0f);
    }
    __syncthreads();
  }

  // ---------------- Z = Os @ Wo^T + bo -> Sf ----------------
  for (int t = wv; t < LqT * 8; t += 8) {
    int tr = t >> 3, tc = t & 7;
    v8f c = {};
#pragma unroll
    for (int k0 = 0; k0 < 128; k0 += 32) {
      v16bf a = load_a_bf(Os, 136, tr * 16, k0, lane);
      v16bf bb = load_b_bf(P.Wo, 128, tc * 16, k0, lane);
      c = WMMA_BF16(a, bb, c);
    }
    store_d_f32(Sf, 132, tr * 16, tc * 16, lane, c, P.bo[tc * 16 + (lane & 15)]);
  }
  __syncthreads();

  // x_mask + residual + LN1 -> Os (bf16)
  for (int q = wv; q < Lq; q += 8) {
    float e[4]; float s = 0.0f;
#pragma unroll
    for (int j = 0; j < 4; ++j) {
      int cc = lane + 32 * j;
      float z = Sf[q * 132 + cc];
      if (P.xmask && q >= qcnt) z = 0.0f;
      z += (float)Xs[q * 136 + cc];
      e[j] = z; s += z;
    }
    s = wave_sum(s);
    float mu = s * (1.0f / 128.0f);
    float v = 0.0f;
#pragma unroll
    for (int j = 0; j < 4; ++j) { float d = e[j] - mu; v += d * d; }
    v = wave_sum(v) * (1.0f / 128.0f);
    float rstd = rsqrtf(v + 1e-5f);
#pragma unroll
    for (int j = 0; j < 4; ++j) {
      int cc = lane + 32 * j;
      Os[q * 136 + cc] = (__bf16)((e[j] - mu) * rstd * P.g1[cc] + P.b1[cc]);
    }
  }
  __syncthreads();

  // rFF: F = Os @ Wl^T + bl -> Sf
  for (int t = wv; t < LqT * 8; t += 8) {
    int tr = t >> 3, tc = t & 7;
    v8f c = {};
#pragma unroll
    for (int k0 = 0; k0 < 128; k0 += 32) {
      v16bf a = load_a_bf(Os, 136, tr * 16, k0, lane);
      v16bf bb = load_b_bf(P.Wl, 128, tc * 16, k0, lane);
      c = WMMA_BF16(a, bb, c);
    }
    store_d_f32(Sf, 132, tr * 16, tc * 16, lane, c, P.bl[tc * 16 + (lane & 15)]);
  }
  __syncthreads();

  // t = u + relu(F); LN2 -> global out (only LqStore rows)
  for (int q = wv; q < P.LqStore; q += 8) {
    float e[4]; float s = 0.0f;
#pragma unroll
    for (int j = 0; j < 4; ++j) {
      int cc = lane + 32 * j;
      float u = (float)Os[q * 136 + cc];
      float t = u + fmaxf(Sf[q * 132 + cc], 0.0f);
      e[j] = t; s += t;
    }
    s = wave_sum(s);
    float mu = s * (1.0f / 128.0f);
    float v = 0.0f;
#pragma unroll
    for (int j = 0; j < 4; ++j) { float d = e[j] - mu; v += d * d; }
    v = wave_sum(v) * (1.0f / 128.0f);
    float rstd = rsqrtf(v + 1e-5f);
#pragma unroll
    for (int j = 0; j < 4; ++j) {
      int cc = lane + 32 * j;
      float val = (e[j] - mu) * rstd * P.g2[cc] + P.b2[cc];
      P.out[(long long)b * P.oBS + (long long)q * C_ + cc] = (__bf16)val;
    }
  }
}

// ---------------- host orchestration ----------------

extern "C" void kernel_launch(void* const* d_in, const int* in_sizes, int n_in,
                              void* d_out, int out_size, void* d_ws, size_t ws_size,
                              hipStream_t stream) {
  (void)in_sizes; (void)n_in; (void)out_size; (void)ws_size;

  const float* x   = (const float*)d_in[0];
  const int* index = (const int*)d_in[1];

  // d_in layout (insertion order of setup_inputs dicts):
  // 0:x 1:index 2:enc0.ind 3..16:enc0.mab1 17..30:enc0.mab2 31:enc1.ind
  // 32..45:enc1.mab1 46..59:enc1.mab2 60:pma.seed 61:pma.Wlin 62:pma.blin
  // 63..76:pma.mab 77..90:dec.mab 91:proj.W 92:proj.b
  // mab array order: Wq Wk Wv bq bk bv Wo bo Wl bl g1 b1 g2 b2
  static const int mabBase[6] = {3, 17, 32, 46, 63, 77};
  static const int woff[5]    = {0, 1, 2, 6, 8};  // Wq Wk Wv Wo Wl

  char* ws = (char*)d_ws;
  auto carve = [&](size_t bytes) -> char* {
    char* p = ws;
    ws += (bytes + 255) & ~(size_t)255;
    return p;
  };

  __bf16* DENSE = (__bf16*)carve((size_t)B_ * LMAX_ * C_ * 2);
  __bf16* HBUF  = (__bf16*)carve((size_t)B_ * M_ * C_ * 2);
  __bf16* YBUF  = (__bf16*)carve((size_t)B_ * LMAX_ * C_ * 2);
  __bf16* OUT1  = (__bf16*)carve((size_t)B_ * C_ * 2);
  int* STARTS   = (int*)carve((size_t)B_ * 4);
  int* COUNTS   = (int*)carve((size_t)B_ * 4);
  __bf16* wbf[6][5];
  for (int m = 0; m < 6; ++m)
    for (int w = 0; w < 5; ++w) wbf[m][w] = (__bf16*)carve((size_t)C_ * C_ * 2);
  __bf16* IND0 = (__bf16*)carve((size_t)M_ * C_ * 2);
  __bf16* IND1 = (__bf16*)carve((size_t)M_ * C_ * 2);
  __bf16* SEED = (__bf16*)carve((size_t)C_ * 2);
  __bf16* WLIN = (__bf16*)carve((size_t)C_ * C_ * 2);
  __bf16* PRJW = (__bf16*)carve((size_t)CO_ * C_ * 2);

  k_starts<<<(B_ + 255) / 256, 256, 0, stream>>>(index, STARTS, COUNTS);
  k_gather<<<(B_ * LMAX_) / 8, 256, 0, stream>>>(x, STARTS, COUNTS, DENSE);

  for (int m = 0; m < 6; ++m)
    for (int w = 0; w < 5; ++w)
      k_cvt<<<(C_ * C_ + 255) / 256, 256, 0, stream>>>((const float*)d_in[mabBase[m] + woff[w]],
                                                       wbf[m][w], C_ * C_);
  k_cvt<<<(M_ * C_ + 255) / 256, 256, 0, stream>>>((const float*)d_in[2], IND0, M_ * C_);
  k_cvt<<<(M_ * C_ + 255) / 256, 256, 0, stream>>>((const float*)d_in[31], IND1, M_ * C_);
  k_cvt<<<1, 256, 0, stream>>>((const float*)d_in[60], SEED, C_);
  k_cvt<<<(C_ * C_ + 255) / 256, 256, 0, stream>>>((const float*)d_in[61], WLIN, C_ * C_);
  k_cvt<<<(CO_ * C_ + 255) / 256, 256, 0, stream>>>((const float*)d_in[91], PRJW, CO_ * C_);

  auto fillMab = [&](int m, MabP& P) {
    int ib = mabBase[m];
    P.Wq = wbf[m][0]; P.Wk = wbf[m][1]; P.Wv = wbf[m][2]; P.Wo = wbf[m][3]; P.Wl = wbf[m][4];
    P.bq = (const float*)d_in[ib + 3];  P.bk = (const float*)d_in[ib + 4];
    P.bv = (const float*)d_in[ib + 5];  P.bo = (const float*)d_in[ib + 7];
    P.bl = (const float*)d_in[ib + 9];
    P.g1 = (const float*)d_in[ib + 10]; P.b1 = (const float*)d_in[ib + 11];
    P.g2 = (const float*)d_in[ib + 12]; P.b2 = (const float*)d_in[ib + 13];
    P.counts = COUNTS;
  };

  const size_t lds = MAB_LDS_BYTES;

  for (int e = 0; e < 2; ++e) {
    MabP P1{}; fillMab(e * 2 + 0, P1);
    P1.X = e ? IND1 : IND0; P1.xBS = 0; P1.Lq = 32; P1.xRows = 32; P1.LqStore = 32;
    P1.Y = DENSE; P1.yBS = (long long)LMAX_ * C_; P1.Lk = 128; P1.yRows = 128;
    P1.use_counts = 1; P1.kfix = 0; P1.xmask = 0;
    P1.out = HBUF; P1.oBS = (long long)M_ * C_;
    k_mab<<<B_, 256, lds, stream>>>(P1);

    MabP P2{}; fillMab(e * 2 + 1, P2);
    P2.X = DENSE; P2.xBS = (long long)LMAX_ * C_; P2.Lq = 128; P2.xRows = 128; P2.LqStore = 128;
    P2.Y = HBUF; P2.yBS = (long long)M_ * C_; P2.Lk = 32; P2.yRows = 32;
    P2.use_counts = 0; P2.kfix = 32; P2.xmask = 1;
    P2.out = DENSE; P2.oBS = (long long)LMAX_ * C_;
    k_mab<<<B_, 256, lds, stream>>>(P2);
  }

  {  // PMA pre-linear: y = relu(dense @ Wlin^T + blin)
    LinP L{};
    L.A = DENSE; L.W = WLIN; L.bias = (const float*)d_in[62];
    L.N = C_; L.relu = 1; L.outf32 = 0; L.out = YBUF;
    k_linear<<<(B_ * LMAX_) / 128, 256, 0, stream>>>(L);
  }
  {  // PMA: MAB(seed, y) with key mask
    MabP P{}; fillMab(4, P);
    P.X = SEED; P.xBS = 0; P.Lq = 16; P.xRows = 1; P.LqStore = 1;
    P.Y = YBUF; P.yBS = (long long)LMAX_ * C_; P.Lk = 128; P.yRows = 128;
    P.use_counts = 1; P.kfix = 0; P.xmask = 0;
    P.out = OUT1; P.oBS = C_;
    k_mab<<<B_, 256, lds, stream>>>(P);
  }
  {  // decoder SAB: MAB(out, out), single token padded to 16q/32k, kcount=1
    MabP P{}; fillMab(5, P);
    P.X = OUT1; P.xBS = C_; P.Lq = 16; P.xRows = 1; P.LqStore = 1;
    P.Y = OUT1; P.yBS = C_; P.Lk = 32; P.yRows = 1;
    P.use_counts = 0; P.kfix = 1; P.xmask = 0;
    P.out = OUT1; P.oBS = C_;
    k_mab<<<B_, 256, lds, stream>>>(P);
  }
  {  // final projection -> f32 d_out [2048, 256]
    LinP L{};
    L.A = OUT1; L.W = PRJW; L.bias = (const float*)d_in[92];
    L.N = CO_; L.relu = 0; L.outf32 = 1; L.out = d_out;
    k_linear<<<B_ / 128, 256, 0, stream>>>(L);
  }
}
